// MambaExpert_24489903522468
// MI455X (gfx1250) — matrix operand order repaired and checked by reading, not compiled, not run on hardware
//
#include <hip/hip_runtime.h>
#include <cstdint>
#include <cstddef>

// ---------------- problem constants ----------------
constexpr int Bsz     = 8;
constexpr int Lseq    = 2048;
constexpr int DMODEL  = 640;
constexpr int DSTATE  = 16;
constexpr int DCONV   = 4;
constexpr int DINNER  = 1280;                 // EXPAND * DMODEL
constexpr int DTRANK  = 40;                   // ceil(640/16)
constexpr int NX      = DTRANK + 2 * DSTATE;  // 72
constexpr int NXPAD   = 128;                  // padded N for W_x^T
constexpr int KDTPAD  = 64;                   // padded K for dt GEMM
constexpr int MROWS   = Bsz * Lseq;           // 16384

// ---------------- WMMA types ----------------
typedef __attribute__((ext_vector_type(16))) __bf16 v16bf;
typedef __attribute__((ext_vector_type(8)))  float  v8f;

union FragBf16 {
    v16bf v;
    uint4 q[2];
};

__device__ __forceinline__ unsigned short f2bf(float f) {
    union { float f; uint32_t u; } cv; cv.f = f;
    uint32_t u = cv.u;
    uint32_t r = u + 0x7FFFu + ((u >> 16) & 1u);   // round to nearest even
    return (unsigned short)(r >> 16);
}

// ---------------- gfx1250 async global->LDS path ----------------
#if defined(__AMDGCN__) &&                                            \
    __has_builtin(__builtin_amdgcn_global_load_async_to_lds_b128) &&  \
    __has_builtin(__builtin_amdgcn_s_wait_asynccnt)
#define USE_ASYNC_LDS 1
typedef int i32x4_t __attribute__((ext_vector_type(4)));
typedef __attribute__((address_space(1))) i32x4_t* gbl_i32x4_p;
typedef __attribute__((address_space(3))) i32x4_t* lds_i32x4_p;
#else
#define USE_ASYNC_LDS 0
#endif

__device__ __forceinline__ void async_wait_le4() {
#if USE_ASYNC_LDS
    __builtin_amdgcn_s_wait_asynccnt(4);
#endif
}
__device__ __forceinline__ void async_wait_0() {
#if USE_ASYNC_LDS
    __builtin_amdgcn_s_wait_asynccnt(0);
#endif
}

__device__ __forceinline__ void copy16_g2l(const unsigned short* gp, unsigned short* lp) {
#if USE_ASYNC_LDS
    __builtin_amdgcn_global_load_async_to_lds_b128(
        (gbl_i32x4_p)(gp), (lds_i32x4_p)(lp), 0, 0);
#else
    *reinterpret_cast<uint4*>(lp) = *reinterpret_cast<const uint4*>(gp);
#endif
}

// ============================================================
// bf16 WMMA GEMM on pre-padded operands:
//   A  : M x Kpad bf16 row-major (Kpad % 32 == 0, M % 128 == 0)
//   Bt : Npad x Kpad bf16 row-major (transposed B, Npad % 128 == 0, zero-padded)
//   C  : M x N fp32 (ldc given, store masked to n < N)
// mode 0: plain store; mode 1: store softplus(val + bias[n])
// Block tile 128x128, BK=32, 256 threads (8 waves, 2x4 wave grid),
// double-buffered LDS staged via GLOBAL_LOAD_ASYNC_TO_LDS_B128.
// ============================================================
#define BM 128
#define BN 128
#define BK 32
#define LDSK 40   // padded K pitch in ushorts (80 B rows -> conflict-free b128 frag loads)

__device__ __forceinline__ void stage_pair(const unsigned short* Ab,
                                           const unsigned short* Bb,
                                           int k0, int ld,
                                           unsigned short* sa, unsigned short* sb,
                                           int tid)
{
    #pragma unroll
    for (int i = 0; i < 2; ++i) {
        const int e   = tid + i * 256;
        const int row = e >> 2;          // 0..127
        const int ch  = (e & 3) << 3;    // 0,8,16,24 (bf16 elements)
        const size_t go = (size_t)row * ld + k0 + ch;
        const int    lo = row * LDSK + ch;
        copy16_g2l(Ab + go, sa + lo);
        copy16_g2l(Bb + go, sb + lo);
    }
}

__global__ __launch_bounds__(256)
void mamba_gemm_bf16_wmma(const unsigned short* __restrict__ A,
                          const unsigned short* __restrict__ Bt,
                          float* __restrict__ C,
                          int N, int Kpad, int ldc,
                          const float* __restrict__ bias, int mode)
{
    __shared__ __align__(16) unsigned short sA[2][BM * LDSK];
    __shared__ __align__(16) unsigned short sB[2][BN * LDSK];

    const int tid  = threadIdx.x;
    const int wave = tid >> 5;
    const int lane = tid & 31;
    const int half = lane >> 4;
    const int l16  = lane & 15;
    const int wm   = wave >> 2;   // 0..1
    const int wn   = wave & 3;    // 0..3
    const int m0   = blockIdx.x * BM;
    const int n0   = blockIdx.y * BN;

    const unsigned short* Ab = A  + (size_t)m0 * Kpad;
    const unsigned short* Bb = Bt + (size_t)n0 * Kpad;

    v8f acc[4][2] = {};

    const int nk = Kpad / BK;
    stage_pair(Ab, Bb, 0, Kpad, sA[0], sB[0], tid);

    for (int t = 0; t < nk; ++t) {
        if (t + 1 < nk) {
            stage_pair(Ab, Bb, (t + 1) * BK, Kpad,
                       sA[(t + 1) & 1], sB[(t + 1) & 1], tid);
            async_wait_le4();     // tile t's 4 copies complete (in-order)
        } else {
            async_wait_0();
        }
        __syncthreads();

        const unsigned short* ca = sA[t & 1];
        const unsigned short* cb = sB[t & 1];

        FragBf16 afr[4], bfr[2];
        #pragma unroll
        for (int i = 0; i < 4; ++i) {
            const int row = wm * 64 + i * 16 + l16;
            // A 16x32 bf16: lanes 0-15 hold K 0..7 & 16..23, lanes 16-31 K 8..15 & 24..31
            afr[i].q[0] = *reinterpret_cast<const uint4*>(&ca[row * LDSK + half * 8]);
            afr[i].q[1] = *reinterpret_cast<const uint4*>(&ca[row * LDSK + 16 + half * 8]);
        }
        #pragma unroll
        for (int j = 0; j < 2; ++j) {
            const int col = wn * 32 + j * 16 + l16;
            // B 32x16 bf16: lane = column; lanes 0-15 K 0..7 & 16..23, lanes 16-31 K 8..15 & 24..31
            bfr[j].q[0] = *reinterpret_cast<const uint4*>(&cb[col * LDSK + half * 8]);
            bfr[j].q[1] = *reinterpret_cast<const uint4*>(&cb[col * LDSK + 16 + half * 8]);
        }

        #pragma unroll
        for (int i = 0; i < 4; ++i)
            #pragma unroll
            for (int j = 0; j < 2; ++j)
                acc[i][j] = __builtin_amdgcn_wmma_f32_16x16x32_bf16(
                    false, afr[i].v, false, bfr[j].v,
                    (short)0, acc[i][j], false, false);

        __syncthreads();
    }

    // ---- store (C layout: N = lane%16, M = vgpr + 8*(lane/16)) ----
    #pragma unroll
    for (int i = 0; i < 4; ++i) {
        #pragma unroll
        for (int j = 0; j < 2; ++j) {
            const int n = n0 + wn * 32 + j * 16 + l16;
            if (n >= N) continue;
            const float bval = (mode == 1) ? bias[n] : 0.f;
            #pragma unroll
            for (int r = 0; r < 8; ++r) {
                const int m = m0 + wm * 64 + i * 16 + r + 8 * half;
                float val = acc[i][j][r];
                if (mode == 1) {
                    float v = val + bval;
                    val = (v > 20.f) ? v : log1pf(__expf(v));   // softplus
                }
                C[(size_t)m * ldc + n] = val;
            }
        }
    }
}

// ============================================================
// Elementwise fp32 -> bf16 convert
// ============================================================
__global__ __launch_bounds__(256)
void cvt_bf16(const float* __restrict__ src, unsigned short* __restrict__ dst, int n)
{
    const int idx = blockIdx.x * 256 + threadIdx.x;
    if (idx < n) dst[idx] = f2bf(src[idx]);
}

// ============================================================
// B (K x N, fp32) -> Bt (Npad x Kpad, bf16), zero-padded
// ============================================================
__global__ __launch_bounds__(256)
void transpose_cvt_bf16(const float* __restrict__ src, unsigned short* __restrict__ dst,
                        int K, int N, int Kpad)
{
    const int idx = blockIdx.x * 256 + threadIdx.x;   // over Npad*Kpad
    const int n = idx / Kpad, k = idx % Kpad;
    float v = (n < N && k < K) ? src[(size_t)k * N + n] : 0.f;
    dst[idx] = f2bf(v);
}

// ============================================================
// dt = dbc[:, :40] -> bf16, K padded to 64 with zeros
// ============================================================
__global__ __launch_bounds__(256)
void dt_pad_cvt(const float* __restrict__ dbc, unsigned short* __restrict__ dst)
{
    const int idx = blockIdx.x * 256 + threadIdx.x;   // over MROWS*64
    const int m = idx >> 6, k = idx & 63;
    float v = (k < DTRANK) ? dbc[(size_t)m * NX + k] : 0.f;
    dst[idx] = f2bf(v);
}

// ============================================================
// Depthwise causal conv (D_CONV=4) + SiLU; writes fp32 uc and bf16 ucbf
// ============================================================
__global__ __launch_bounds__(256)
void mamba_conv_silu(const float* __restrict__ uz,
                     const float* __restrict__ cw,
                     const float* __restrict__ cb,
                     float* __restrict__ uc,
                     unsigned short* __restrict__ ucbf)
{
    const size_t idx = (size_t)blockIdx.x * 256 + threadIdx.x;
    if (idx >= (size_t)MROWS * DINNER) return;
    const int    d  = (int)(idx % DINNER);
    const size_t bl = idx / DINNER;            // b*L + l
    const int    l  = (int)(bl % Lseq);

    float acc = cb[d];
    #pragma unroll
    for (int k = 0; k < DCONV; ++k) {
        const int lk = l + k - (DCONV - 1);
        if (lk >= 0)
            acc += cw[d * DCONV + k] * uz[(bl + (k - (DCONV - 1))) * (2 * DINNER) + d];
    }
    const float s = acc / (1.f + __expf(-acc));   // SiLU
    uc[bl * DINNER + d]   = s;
    ucbf[bl * DINNER + d] = f2bf(s);
}

// ============================================================
// Selective scan: one lane per (b, d) channel; h[16] in registers.
// B/C vectors (dbc cols 40..71) staged per 64-step chunk into LDS.
// Fused epilogue: y = (y + D*uc) * silu(z). y aliases uc.
// ============================================================
#define SCHUNK 64
__global__ __launch_bounds__(256)
void mamba_scan(const float* __restrict__ delta,
                const float* uc,                    // no restrict: y aliases uc
                const float* __restrict__ dbc,
                const float* __restrict__ uz,       // z at cols 1280..2559
                const float* __restrict__ A_log,
                const float* __restrict__ Dp,
                float* y)
{
    __shared__ float sbc[SCHUNK][32];
    const int b = blockIdx.x / (DINNER / 256);
    const int d = (blockIdx.x % (DINNER / 256)) * 256 + threadIdx.x;

    float Arow[DSTATE], h[DSTATE];
    #pragma unroll
    for (int s = 0; s < DSTATE; ++s) {
        Arow[s] = -__expf(A_log[d * DSTATE + s]);
        h[s] = 0.f;
    }
    const float Dd = Dp[d];
    const size_t base = (size_t)b * Lseq;

    for (int t0 = 0; t0 < Lseq; t0 += SCHUNK) {
        #pragma unroll
        for (int i = 0; i < (SCHUNK * 32) / 256; ++i) {
            const int e  = threadIdx.x + i * 256;
            const int tl = e >> 5, s = e & 31;
            sbc[tl][s] = dbc[(base + t0 + tl) * NX + DTRANK + s];
        }
        __syncthreads();

        for (int tl = 0; tl < SCHUNK; ++tl) {
            const size_t row = base + t0 + tl;
            const float dlt = delta[row * DINNER + d];
            const float u_t = uc[row * DINNER + d];
            const float z_t = uz[row * (2 * DINNER) + DINNER + d];
            const float du  = dlt * u_t;
            float yv = 0.f;
            #pragma unroll
            for (int s = 0; s < DSTATE; ++s) {
                const float dA = __expf(dlt * Arow[s]);
                h[s] = dA * h[s] + du * sbc[tl][s];
                yv += h[s] * sbc[tl][DSTATE + s];
            }
            yv = (yv + Dd * u_t) * (z_t / (1.f + __expf(-z_t)));
            y[row * DINNER + d] = yv;
        }
        __syncthreads();
    }
}

// ============================================================
// ybar[b][d] = mean_l y[b,l,d]   (linearity: mean before W_out)
// ============================================================
__global__ __launch_bounds__(256)
void mamba_colmean(const float* __restrict__ y, float* __restrict__ ybar)
{
    const int idx = blockIdx.x * 256 + threadIdx.x;   // 0..10239
    const int b = idx / DINNER, d = idx % DINNER;
    const float* p = y + ((size_t)b * Lseq) * DINNER + d;
    float acc = 0.f;
    for (int l = 0; l < Lseq; ++l) acc += p[(size_t)l * DINNER];
    ybar[idx] = acc * (1.f / Lseq);
}

// ============================================================
// out[b][n] = sum_d ybar[b,d] * W_out[d,n]   (tiny GEMM, fp32)
// ============================================================
__global__ __launch_bounds__(256)
void mamba_out(const float* __restrict__ ybar,
               const float* __restrict__ W_out,
               float* __restrict__ out)
{
    const int idx = blockIdx.x * 256 + threadIdx.x;   // 0..5119
    const int b = idx / DMODEL, n = idx % DMODEL;
    float acc = 0.f;
    for (int d = 0; d < DINNER; ++d)
        acc += ybar[b * DINNER + d] * W_out[(size_t)d * DMODEL + n];
    out[idx] = acc;
}

// ============================================================
extern "C" void kernel_launch(void* const* d_in, const int* in_sizes, int n_in,
                              void* d_out, int out_size, void* d_ws, size_t ws_size,
                              hipStream_t stream)
{
    const float* x      = (const float*)d_in[0];
    const float* W_in   = (const float*)d_in[1];
    const float* conv_w = (const float*)d_in[2];
    const float* conv_b = (const float*)d_in[3];
    const float* W_x    = (const float*)d_in[4];
    const float* W_dt   = (const float*)d_in[5];
    const float* b_dt   = (const float*)d_in[6];
    const float* A_log  = (const float*)d_in[7];
    const float* Dp     = (const float*)d_in[8];
    const float* W_out  = (const float*)d_in[9];
    float* out = (float*)d_out;

    char* ws = (char*)d_ws;
    size_t off = 0;
    auto wsalloc = [&](size_t bytes) -> void* {
        void* p = ws + off;
        off = (off + bytes + 255) & ~(size_t)255;
        return p;
    };
    float*          uz    = (float*)wsalloc((size_t)MROWS * 2 * DINNER * 4);   // 168 MB
    float*          uc    = (float*)wsalloc((size_t)MROWS * DINNER * 4);       //  84 MB
    float*          dbc   = (float*)wsalloc((size_t)MROWS * NX * 4);           //   5 MB
    float*          delta = (float*)wsalloc((size_t)MROWS * DINNER * 4);       //  84 MB
    float*          ybar  = (float*)wsalloc((size_t)Bsz * DINNER * 4);
    unsigned short* xbf   = (unsigned short*)wsalloc((size_t)MROWS * DMODEL * 2); // 21 MB
    unsigned short* WinT  = (unsigned short*)wsalloc((size_t)2 * DINNER * DMODEL * 2);
    unsigned short* ucbf  = (unsigned short*)wsalloc((size_t)MROWS * DINNER * 2); // 42 MB
    unsigned short* WxT   = (unsigned short*)wsalloc((size_t)NXPAD * DINNER * 2);
    unsigned short* dtbf  = (unsigned short*)wsalloc((size_t)MROWS * KDTPAD * 2); // 2 MB
    unsigned short* WdtT  = (unsigned short*)wsalloc((size_t)DINNER * KDTPAD * 2);
    float* y = uc;   // in-place: each (b,t,d) read before written by same thread

    // ---- operand prep (bf16, padded, B transposed) ----
    cvt_bf16<<<(MROWS * DMODEL) / 256, 256, 0, stream>>>(x, xbf, MROWS * DMODEL);
    transpose_cvt_bf16<<<(2 * DINNER * DMODEL) / 256, 256, 0, stream>>>(
        W_in, WinT, DMODEL, 2 * DINNER, DMODEL);
    transpose_cvt_bf16<<<(NXPAD * DINNER) / 256, 256, 0, stream>>>(
        W_x, WxT, DINNER, NX, DINNER);
    transpose_cvt_bf16<<<(DINNER * KDTPAD) / 256, 256, 0, stream>>>(
        W_dt, WdtT, DTRANK, DINNER, KDTPAD);

    // 1) uz = x @ W_in  (M=16384, N=2560, K=640)
    mamba_gemm_bf16_wmma<<<dim3(MROWS / BM, (2 * DINNER) / BN), 256, 0, stream>>>(
        xbf, WinT, uz, 2 * DINNER, DMODEL, 2 * DINNER, nullptr, 0);

    // 2) uc = silu(causal_conv(u) + b)  (+ bf16 copy)
    mamba_conv_silu<<<((size_t)MROWS * DINNER + 255) / 256, 256, 0, stream>>>(
        uz, conv_w, conv_b, uc, ucbf);

    // 3) dbc = uc @ W_x  (N=72 via padded Bt, masked store)
    mamba_gemm_bf16_wmma<<<dim3(MROWS / BM, 1), 256, 0, stream>>>(
        ucbf, WxT, dbc, NX, DINNER, NX, nullptr, 0);

    // 3b) dt slice -> bf16, K padded to 64
    dt_pad_cvt<<<(MROWS * KDTPAD) / 256, 256, 0, stream>>>(dbc, dtbf);

    // 4) delta = softplus(dt @ W_dt + b_dt)
    mamba_gemm_bf16_wmma<<<dim3(MROWS / BM, DINNER / BN), 256, 0, stream>>>(
        dtbf, WdtT, delta, DINNER, KDTPAD, DINNER, b_dt, 1);

    // 5) selective scan + gating epilogue (y in-place over uc)
    mamba_scan<<<Bsz * (DINNER / 256), 256, 0, stream>>>(
        delta, uc, dbc, uz, A_log, Dp, y);

    // 6) ybar = mean_l y
    mamba_colmean<<<(Bsz * DINNER) / 256, 256, 0, stream>>>(y, ybar);

    // 7) out = ybar @ W_out
    mamba_out<<<(Bsz * DMODEL) / 256, 256, 0, stream>>>(ybar, W_out, out);
}
